// Prot_EGNN_28166395527436
// MI455X (gfx1250) — matrix operand order repaired and checked by reading, not compiled, not run on hardware
//
#include <hip/hip_runtime.h>

// ---------------- model constants ----------------
#define DD   64      // node channels
#define MM   16      // msg channels
#define EINK 129     // edge mlp input dim
#define K1P  160     // EINK padded to mult of 32 (5 k-tiles)
#define EH_  258     // edge hidden
#define EHP  288     // EH padded: 18 n-tiles of 16 / 9 k-tiles of 32
#define CHH  64      // coord mlp hidden
#define KCP  32      // coord mlp K (16) padded to 32
#define NHH  128     // node hidden
#define KN1  80      // D + M
#define KN1P 96      // padded to mult of 32 (3 k-tiles)
#define TE   64      // edges per block
#define TN   64      // nodes per block

typedef __attribute__((ext_vector_type(16))) _Float16 v16h;
typedef __attribute__((ext_vector_type(8)))  float    v8f;
typedef __attribute__((ext_vector_type(2)))  _Float16 h2;

union PK { unsigned u; h2 h; };

__device__ __forceinline__ float silu_f(float v){ return v / (1.0f + __expf(-v)); }

__device__ __forceinline__ v8f wmma16(v16h a, v16h b, v8f c){
  // v_wmma_f32_16x16x32_f16 : D = A(16x32 f16) * B(32x16 f16) + C(16x16 f32)
  return __builtin_amdgcn_wmma_f32_16x16x32_f16(false, a, false, b, (short)0, c, false, false);
}

// A fragment (16x32 f16) from LDS, row-major rows with even stride (in halves).
// lane L: row m = L&15 ; K-half = (L>>4)*8 ; vgpr j holds pair:
//   k = ktile*32 + (j>>2)*16 + (L>>4)*8 + (j&3)*2
__device__ __forceinline__ v16h frag_A_lds(const _Float16* base, int row0, int ktile,
                                           int stride, int lane){
  const unsigned* p = (const unsigned*)base;
  int m = lane & 15, hf = lane >> 4;
  v16h a;
  #pragma unroll
  for (int j = 0; j < 8; ++j){
    int k = ktile*32 + (j>>2)*16 + hf*8 + (j&3)*2;
    PK q; q.u = p[((row0 + m)*stride + k) >> 1];
    a[2*j] = q.h[0]; a[2*j+1] = q.h[1];
  }
  return a;
}

// B fragment (32x16 f16) from pre-transposed global weights WT[n][k] (stride in halves).
// lane L: col n = L&15 ; K-half = (L>>4)*8 ; vgpr j holds k = ktile*32 + (j>>2)*16 + (L>>4)*8 + (j&3)*2
__device__ __forceinline__ v16h frag_B_gl(const _Float16* __restrict__ wt, int ncol0,
                                          int ktile, int stride, int lane){
  const unsigned* p = (const unsigned*)wt;
  int n = lane & 15, hf = lane >> 4;
  v16h b;
  #pragma unroll
  for (int j = 0; j < 8; ++j){
    int k = ktile*32 + (j>>2)*16 + hf*8 + (j&3)*2;
    PK q; q.u = p[((ncol0 + n)*stride + k) >> 1];
    b[2*j] = q.h[0]; b[2*j+1] = q.h[1];
  }
  return b;
}

__device__ __forceinline__ v8f vzero8(){
  v8f z = {0.f,0.f,0.f,0.f,0.f,0.f,0.f,0.f};
  return z;
}

// ---------------- init: x = emb[feats] (f32 + f16 mirror), pos = coors ----------------
__global__ __launch_bounds__(256) void egnn_init(const int* __restrict__ feats,
                                                 const float* __restrict__ emb,
                                                 const float* __restrict__ coors,
                                                 float* __restrict__ x,
                                                 _Float16* __restrict__ x16,
                                                 float* __restrict__ pos, int Nn){
  int tid = blockIdx.x * 256 + threadIdx.x;
  if (tid < Nn * DD){
    int i = tid >> 6, d = tid & 63;
    float v = emb[feats[i]*DD + d];
    x[tid]   = v;
    x16[tid] = (_Float16)v;
  }
  if (tid < Nn * 3) pos[tid] = coors[tid];
}

// ---------------- zero aggregation buffers ----------------
__global__ __launch_bounds__(256) void egnn_zero(float* __restrict__ agg_msg,
                                                 float* __restrict__ agg_pos, int Nn){
  int tid = blockIdx.x * 256 + threadIdx.x;
  if (tid < Nn * MM) agg_msg[tid] = 0.f;
  if (tid < Nn * 3)  agg_pos[tid] = 0.f;
}

// ---------------- weight prep: f32 [K][N] -> f16 WT[n][kpad] (zero padded) ----------------
__global__ __launch_bounds__(256) void egnn_prep(const float* __restrict__ We1,
                                                 const float* __restrict__ We2,
                                                 const float* __restrict__ Wn1,
                                                 const float* __restrict__ Wn2,
                                                 const float* __restrict__ Wc1,
                                                 _Float16* __restrict__ WT1,
                                                 _Float16* __restrict__ WT2,
                                                 _Float16* __restrict__ WTn1,
                                                 _Float16* __restrict__ WTn2,
                                                 _Float16* __restrict__ WTc1){
  const int S1 = 3*EHP*K1P, S2 = 3*MM*EHP, S3 = 3*NHH*KN1P, S4 = 3*DD*NHH, S5 = 3*CHH*KCP;
  int tid = blockIdx.x * 256 + threadIdx.x;
  if (tid < S1){
    int l = tid / (EHP*K1P), r = tid % (EHP*K1P);
    int n = r / K1P, k = r % K1P;
    float v = (n < EH_ && k < EINK) ? We1[(l*EINK + k)*EH_ + n] : 0.f;
    WT1[tid] = (_Float16)v;
  } else if (tid < S1 + S2){
    int t = tid - S1;
    int l = t / (MM*EHP), r = t % (MM*EHP);
    int n = r / EHP, k = r % EHP;
    float v = (k < EH_) ? We2[(l*EH_ + k)*MM + n] : 0.f;
    WT2[t] = (_Float16)v;
  } else if (tid < S1 + S2 + S3){
    int t = tid - S1 - S2;
    int l = t / (NHH*KN1P), r = t % (NHH*KN1P);
    int n = r / KN1P, k = r % KN1P;
    float v = (k < KN1) ? Wn1[(l*KN1 + k)*NHH + n] : 0.f;
    WTn1[t] = (_Float16)v;
  } else if (tid < S1 + S2 + S3 + S4){
    int t = tid - S1 - S2 - S3;
    int l = t / (DD*NHH), r = t % (DD*NHH);
    int n = r / NHH, k = r % NHH;
    WTn2[t] = (_Float16)Wn2[(l*NHH + k)*DD + n];
  } else if (tid < S1 + S2 + S3 + S4 + S5){
    int t = tid - S1 - S2 - S3 - S4;
    int l = t / (CHH*KCP), r = t % (CHH*KCP);
    int n = r / KCP, k = r % KCP;
    float v = (k < MM) ? Wc1[(l*MM + k)*CHH + n] : 0.f;
    WTc1[t] = (_Float16)v;
  }
}

// ---------------- edge kernel: message + coord weight + scatter ----------------
// LDS carve (64000 B total):
//   [0      , 20480) : Abuf (m_in f16 64x160)  -- phases build/GEMM1; aliased by SBuf later
//   [0      , 16384) : SBuf (coord f32 64x64)  -- phase coord (alias of Abuf)
//   [20480  , 57344) : Hbuf (hidden f16 64x288)
//   [57344  , 61440) : MsgH (msg f16 64x32, cols16..31 zero)
//   [61440  , 62464) : RelS (f32 64x4)
//   [62464  , 63488) : PartS(f32 64x4)
//   [63488  , 63744) : CwS  (f32 64)
//   [63744  , 64000) : TgtS (int 64)
__global__ __launch_bounds__(256, 1) void egnn_edge(
    const _Float16* __restrict__ x16, const float* __restrict__ pos,
    const int* __restrict__ esrc, const int* __restrict__ etgt,
    const _Float16* __restrict__ WT1, const float* __restrict__ be1,
    const _Float16* __restrict__ WT2, const float* __restrict__ be2,
    const _Float16* __restrict__ WTc1, const float* __restrict__ bc1,
    const float* __restrict__ Wc2, const float* __restrict__ bc2,
    float* __restrict__ agg_msg, float* __restrict__ agg_pos, int E)
{
  __shared__ __align__(16) char SM[64000];
  _Float16* Abuf = (_Float16*)(SM);
  float*    SBuf = (float*)   (SM);            // alias, used after Abuf is dead
  _Float16* Hbuf = (_Float16*)(SM + 20480);
  _Float16* MsgH = (_Float16*)(SM + 57344);
  float*    RelS = (float*)   (SM + 61440);
  float*    PartS= (float*)   (SM + 62464);
  float*    CwS  = (float*)   (SM + 63488);
  int*      TgtS = (int*)     (SM + 63744);

  const int tid  = threadIdx.x;
  const int lane = tid & 31;
  const int wave = tid >> 5;
  const int e0   = blockIdx.x * TE;

  // ---- build A = [x_tgt(64) | x_src(64) | d2 | 0...] from f16 mirror ----
  {
    int e = tid & 63, part = tid >> 6;         // part: 0,1 -> tgt halves; 2,3 -> src halves
    int ge = e0 + e;
    bool vld = ge < E;
    int node = vld ? ((part < 2) ? etgt[ge] : esrc[ge]) : 0;
    int cbase = (part & 1) * 32;               // 32 f16 channels = 64 B = 4x uint4
    const uint4* xp = (const uint4*)(x16 + (size_t)node * DD + cbase);
    uint4* ap = (uint4*)(Abuf + e * K1P + (part >> 1) * DD + cbase);
    uint4 zz = {0u,0u,0u,0u};
    #pragma unroll
    for (int i = 0; i < 4; ++i) ap[i] = vld ? xp[i] : zz;
  }
  if (tid < TE){
    int e = tid, ge = e0 + e;
    bool vld = ge < E;
    int t = vld ? etgt[ge] : 0, s = vld ? esrc[ge] : 0;
    TgtS[e] = t;
    float rx = vld ? (pos[t*3+0] - pos[s*3+0]) : 0.f;
    float ry = vld ? (pos[t*3+1] - pos[s*3+1]) : 0.f;
    float rz = vld ? (pos[t*3+2] - pos[s*3+2]) : 0.f;
    float d2 = rx*rx + ry*ry + rz*rz;
    RelS[e*4+0] = rx; RelS[e*4+1] = ry; RelS[e*4+2] = rz;
    PK p; p.h[0] = (_Float16)d2; p.h[1] = (_Float16)0.f;
    *(unsigned*)(Abuf + e*K1P + 128) = p.u;            // cols 128 (d2), 129 (pad)
    #pragma unroll
    for (int c = 130; c < K1P; c += 2)                 // zero pad cols 130..159
      *(unsigned*)(Abuf + e*K1P + c) = 0u;
  }
  __syncthreads();

  // ---- GEMM1: [64 x 160] x WT1^T -> hidden [64 x 288] (cols>=258 are zero) ----
  {
    int mtile = wave & 3;           // 4 row tiles of 16 edges
    int nh    = wave >> 2;          // two halves of 18 n-tiles
    int nt0   = nh * 9;
    v8f acc[9];
    #pragma unroll
    for (int i = 0; i < 9; ++i) acc[i] = vzero8();
    #pragma unroll
    for (int kt = 0; kt < 5; ++kt){
      v16h a = frag_A_lds(Abuf, mtile*16, kt, K1P, lane);
      #pragma unroll
      for (int nt = 0; nt < 9; ++nt){
        v16h b = frag_B_gl(WT1, (nt0 + nt)*16, kt, K1P, lane);
        acc[nt] = wmma16(a, b, acc[nt]);
      }
    }
    int nn = lane & 15, mh = lane >> 4;
    #pragma unroll
    for (int nt = 0; nt < 9; ++nt){
      int ch = (nt0 + nt)*16 + nn;
      float bias = (ch < EH_) ? be1[ch] : 0.f;
      #pragma unroll
      for (int i = 0; i < 8; ++i){
        int e = mtile*16 + mh*8 + i;
        Hbuf[e*EHP + ch] = (_Float16)silu_f(acc[nt][i] + bias);
      }
    }
  }
  __syncthreads();

  // ---- GEMM2: [64 x 288] x WT2^T -> msg [64 x 16] (stored f16, A-ready, padded) ----
  if (wave < 4){
    int mtile = wave;
    v8f acc = vzero8();
    #pragma unroll
    for (int kt = 0; kt < 9; ++kt){
      v16h a = frag_A_lds(Hbuf, mtile*16, kt, EHP, lane);
      v16h b = frag_B_gl(WT2, 0, kt, EHP, lane);
      acc = wmma16(a, b, acc);
    }
    int nn = lane & 15, mh = lane >> 4;
    float bias = be2[nn];
    #pragma unroll
    for (int i = 0; i < 8; ++i){
      int e = mtile*16 + mh*8 + i;
      MsgH[e*KCP + nn]      = (_Float16)silu_f(acc[i] + bias);
      MsgH[e*KCP + 16 + nn] = (_Float16)0.f;   // K pad for coord GEMM
    }
  }
  __syncthreads();

  // ---- coord GEMM: msg[64x32pad] x WTc1^T -> silu(.+bc1)*Wc2 -> SBuf [64x64] ----
  {
    int mtile = wave & 3, nh = wave >> 2;       // 2 n-tiles per wave
    v8f acc[2];
    #pragma unroll
    for (int i = 0; i < 2; ++i) acc[i] = vzero8();
    #pragma unroll
    for (int nt = 0; nt < 2; ++nt){
      v16h a = frag_A_lds(MsgH, mtile*16, 0, KCP, lane);
      v16h b = frag_B_gl(WTc1, (nh*2 + nt)*16, 0, KCP, lane);
      acc[nt] = wmma16(a, b, acc[nt]);
    }
    int nn = lane & 15, mh = lane >> 4;
    #pragma unroll
    for (int nt = 0; nt < 2; ++nt){
      int ch = (nh*2 + nt)*16 + nn;
      float b1 = bc1[ch], w2 = Wc2[ch];
      #pragma unroll
      for (int i = 0; i < 8; ++i){
        int e = mtile*16 + mh*8 + i;
        SBuf[e*CHH + ch] = silu_f(acc[nt][i] + b1) * w2;
      }
    }
  }
  __syncthreads();

  // ---- row-sum of SBuf -> cw ----
  {
    int e = tid >> 2, q = tid & 3;
    float p = 0.f;
    #pragma unroll
    for (int i = 0; i < 16; ++i) p += SBuf[e*CHH + q*16 + i];
    PartS[e*4 + q] = p;
  }
  __syncthreads();
  if (tid < TE)
    CwS[tid] = PartS[tid*4] + PartS[tid*4+1] + PartS[tid*4+2] + PartS[tid*4+3] + bc2[0];
  __syncthreads();

  // ---- scatter-add msg and rel*cw to target nodes ----
  #pragma unroll
  for (int it = 0; it < 4; ++it){
    int idx = it*256 + tid;
    int e = idx >> 4, c = idx & 15;
    if (e0 + e < E)
      atomicAdd(&agg_msg[(size_t)TgtS[e]*MM + c], (float)MsgH[e*KCP + c]);
  }
  if (tid < TE*3){
    int e = tid / 3, c = tid - e*3;
    if (e0 + e < E)
      atomicAdd(&agg_pos[(size_t)TgtS[e]*3 + c], RelS[e*4+c] * CwS[e]);
  }
}

// ---------------- node kernel: node MLP + residuals ----------------
__global__ __launch_bounds__(256, 1) void egnn_node(
    float* __restrict__ x, _Float16* __restrict__ x16, float* __restrict__ pos,
    const float* __restrict__ agg_msg, const float* __restrict__ agg_pos,
    const _Float16* __restrict__ WTn1, const float* __restrict__ bn1,
    const _Float16* __restrict__ WTn2, const float* __restrict__ bn2, int Nn)
{
  __shared__ __align__(16) _Float16 Abuf[TN * KN1P];  // [x | agg_msg | 0]
  __shared__ __align__(16) _Float16 Hbuf[TN * NHH];

  const int tid  = threadIdx.x;
  const int lane = tid & 31;
  const int wave = tid >> 5;
  const int n0   = blockIdx.x * TN;

  // ---- build A = [x(64) | agg_msg(16) | zeros(16)] ----
  {
    int r = tid & 63, part = tid >> 6;       // 24 cols per part
    int node = n0 + r;
    bool ok = node < Nn;
    _Float16* arow = Abuf + r * KN1P;
    #pragma unroll
    for (int c = part*24; c < part*24 + 24; c += 2){
      float f0 = 0.f, f1 = 0.f;
      if (ok){
        f0 = (c   < DD) ? x[(size_t)node*DD + c]   : (c   < KN1 ? agg_msg[(size_t)node*MM + (c   - DD)] : 0.f);
        f1 = (c+1 < DD) ? x[(size_t)node*DD + c+1] : (c+1 < KN1 ? agg_msg[(size_t)node*MM + (c+1 - DD)] : 0.f);
      }
      PK p; p.h[0] = (_Float16)f0; p.h[1] = (_Float16)f1;
      *(unsigned*)(arow + c) = p.u;
    }
  }
  __syncthreads();

  // ---- MLP1: [64 x 96] x WTn1^T -> [64 x 128] ----
  {
    int mtile = wave & 3, nh = wave >> 2;    // 4 n-tiles per wave
    v8f acc[4];
    #pragma unroll
    for (int i = 0; i < 4; ++i) acc[i] = vzero8();
    #pragma unroll
    for (int kt = 0; kt < 3; ++kt){
      v16h a = frag_A_lds(Abuf, mtile*16, kt, KN1P, lane);
      #pragma unroll
      for (int nt = 0; nt < 4; ++nt){
        v16h b = frag_B_gl(WTn1, (nh*4 + nt)*16, kt, KN1P, lane);
        acc[nt] = wmma16(a, b, acc[nt]);
      }
    }
    int nn = lane & 15, mh = lane >> 4;
    #pragma unroll
    for (int nt = 0; nt < 4; ++nt){
      int ch = (nh*4 + nt)*16 + nn;
      float bias = bn1[ch];
      #pragma unroll
      for (int i = 0; i < 8; ++i){
        int r = mtile*16 + mh*8 + i;
        Hbuf[r*NHH + ch] = (_Float16)silu_f(acc[nt][i] + bias);
      }
    }
  }
  __syncthreads();

  // ---- MLP2: [64 x 128] x WTn2^T -> [64 x 64]; x += result (f32 + f16 mirror) ----
  {
    int mtile = wave & 3, nh = wave >> 2;    // 2 n-tiles per wave
    v8f acc[2];
    #pragma unroll
    for (int i = 0; i < 2; ++i) acc[i] = vzero8();
    #pragma unroll
    for (int kt = 0; kt < 4; ++kt){
      v16h a = frag_A_lds(Hbuf, mtile*16, kt, NHH, lane);
      #pragma unroll
      for (int nt = 0; nt < 2; ++nt){
        v16h b = frag_B_gl(WTn2, (nh*2 + nt)*16, kt, NHH, lane);
        acc[nt] = wmma16(a, b, acc[nt]);
      }
    }
    int nn = lane & 15, mh = lane >> 4;
    #pragma unroll
    for (int nt = 0; nt < 2; ++nt){
      int ch = (nh*2 + nt)*16 + nn;
      float bias = bn2[ch];
      #pragma unroll
      for (int i = 0; i < 8; ++i){
        int r = mtile*16 + mh*8 + i;
        int node = n0 + r;
        if (node < Nn){
          size_t ix = (size_t)node*DD + ch;
          float nv = x[ix] + acc[nt][i] + bias;
          x[ix]   = nv;
          x16[ix] = (_Float16)nv;
        }
      }
    }
  }

  // ---- pos += agg_pos ----
  if (tid < TN*3){
    int r = tid / 3, c = tid - r*3;
    int node = n0 + r;
    if (node < Nn) pos[(size_t)node*3 + c] += agg_pos[(size_t)node*3 + c];
  }
}

// ---------------- host launch ----------------
extern "C" void kernel_launch(void* const* d_in, const int* in_sizes, int n_in,
                              void* d_out, int out_size, void* d_ws, size_t ws_size,
                              hipStream_t stream)
{
  const int*   feats = (const int*)  d_in[0];
  const float* coors = (const float*)d_in[1];
  const int*   ei    = (const int*)  d_in[2];
  const float* emb   = (const float*)d_in[3];
  const float* We1 = (const float*)d_in[4];  const float* be1 = (const float*)d_in[5];
  const float* We2 = (const float*)d_in[6];  const float* be2 = (const float*)d_in[7];
  const float* Wc1 = (const float*)d_in[8];  const float* bc1 = (const float*)d_in[9];
  const float* Wc2 = (const float*)d_in[10]; const float* bc2 = (const float*)d_in[11];
  const float* Wn1 = (const float*)d_in[12]; const float* bn1 = (const float*)d_in[13];
  const float* Wn2 = (const float*)d_in[14]; const float* bn2 = (const float*)d_in[15];

  const int Nn = in_sizes[0];
  const int E  = in_sizes[2] / 2;
  const int* esrc = ei;        // edge_index[0]
  const int* etgt = ei + E;    // edge_index[1]

  float* x   = (float*)d_out;                  // [N, 64] flat
  float* pos = x + (size_t)Nn * DD;            // [N, 3]  flat

  // workspace carve-out
  char* ws = (char*)d_ws;
  size_t off = 0;
  auto take = [&](size_t bytes) -> char* {
    char* p = ws + off;
    off += (bytes + 255) & ~(size_t)255;
    return p;
  };
  float*    agg_msg = (float*)   take((size_t)Nn * MM * 4);
  float*    agg_pos = (float*)   take((size_t)Nn * 3  * 4);
  _Float16* x16     = (_Float16*)take((size_t)Nn * DD * 2);
  _Float16* WT1     = (_Float16*)take((size_t)3 * EHP * K1P * 2);
  _Float16* WT2     = (_Float16*)take((size_t)3 * MM  * EHP * 2);
  _Float16* WTn1    = (_Float16*)take((size_t)3 * NHH * KN1P * 2);
  _Float16* WTn2    = (_Float16*)take((size_t)3 * DD  * NHH * 2);
  _Float16* WTc1    = (_Float16*)take((size_t)3 * CHH * KCP * 2);

  egnn_init<<<dim3((Nn*DD + 255)/256), dim3(256), 0, stream>>>(
      feats, emb, coors, x, x16, pos, Nn);

  const int prepTot = 3*EHP*K1P + 3*MM*EHP + 3*NHH*KN1P + 3*DD*NHH + 3*CHH*KCP;
  egnn_prep<<<dim3((prepTot + 255)/256), dim3(256), 0, stream>>>(
      We1, We2, Wn1, Wn2, Wc1, WT1, WT2, WTn1, WTn2, WTc1);

  for (int l = 0; l < 3; ++l){
    egnn_zero<<<dim3((Nn*MM + 255)/256), dim3(256), 0, stream>>>(agg_msg, agg_pos, Nn);
    egnn_edge<<<dim3((E + TE - 1)/TE), dim3(256), 0, stream>>>(
        x16, pos, esrc, etgt,
        WT1 + (size_t)l*EHP*K1P, be1 + (size_t)l*EH_,
        WT2 + (size_t)l*MM*EHP,  be2 + (size_t)l*MM,
        WTc1 + (size_t)l*CHH*KCP, bc1 + (size_t)l*CHH,
        Wc2 + (size_t)l*CHH,     bc2 + l,
        agg_msg, agg_pos, E);
    egnn_node<<<dim3((Nn + TN - 1)/TN), dim3(256), 0, stream>>>(
        x, x16, pos, agg_msg, agg_pos,
        WTn1 + (size_t)l*NHH*KN1P, bn1 + (size_t)l*NHH,
        WTn2 + (size_t)l*DD*NHH,   bn2 + (size_t)l*DD, Nn);
  }
}
